// GNNEncoder_65936337928868
// MI455X (gfx1250) — matrix operand order repaired and checked by reading, not compiled, not run on hardware
//
#include <hip/hip_runtime.h>

// ---------------------------------------------------------------------------
// GIN encoder for MI455X (gfx1250, wave32).
// GEMMs: v_wmma_f32_16x16x32_bf16, one wave owns a 16x64 strip (4 N-tiles
//        share one A fragment -> 4 WMMA per 10 b128 loads).
// Edge phase: float4 gathers + global_atomic_add_f32 scatter (HBM-bound).
// ---------------------------------------------------------------------------

#define NN 50000
#define EE 250000
#define DD 300
#define LL 5
#define D2 600
#define KP1 320   // GEMM1 K  (D padded to mult of 32)
#define NP1 640   // GEMM1 out cols (2D padded to mult of 64)
#define KP2 640   // GEMM2 K  (= NP1, mult of 32)
#define NP2 320   // GEMM2 out cols (D padded to mult of 64)

typedef __attribute__((ext_vector_type(16))) __bf16 v16bf;
typedef __attribute__((ext_vector_type(8)))  __bf16 v8bf;
typedef __attribute__((ext_vector_type(8)))  float  v8f;
typedef __attribute__((ext_vector_type(4)))  float  vf4;
typedef __attribute__((ext_vector_type(4)))  unsigned short vu4;

__device__ __forceinline__ unsigned short f2bf(float f) {
  unsigned int u = __builtin_bit_cast(unsigned int, f);
  unsigned int r = u + 0x7FFFu + ((u >> 16) & 1u);   // round-to-nearest-even
  return (unsigned short)(r >> 16);
}
__device__ __forceinline__ float bf2f(unsigned short h) {
  unsigned int u = ((unsigned int)h) << 16;
  return __builtin_bit_cast(float, u);
}

// ---- AtomEncoder (float4): h[n,d..d+3] = sum_c atom_emb[c, x[n,c], d..] ----
__global__ void atom_encoder_kernel(const int* __restrict__ x,
                                    const float* __restrict__ atom_emb,
                                    float* __restrict__ h) {
  int idx = blockIdx.x * blockDim.x + threadIdx.x;
  if (idx >= NN * (DD / 4)) return;
  int n = idx / (DD / 4), d4 = idx % (DD / 4);
  vf4 acc = {0.f, 0.f, 0.f, 0.f};
#pragma unroll
  for (int c = 0; c < 9; ++c) {
    int code = x[n * 9 + c];
    vf4 v = *(const vf4*)(atom_emb + ((size_t)(c * 128 + code)) * DD + d4 * 4);
    acc += v;
  }
  *(vf4*)(h + (size_t)n * DD + d4 * 4) = acc;
}

// ---- Edge phase (float4): msg = relu(h[src]+bond), atomic scatter ----------
__global__ void edge_kernel(const int* __restrict__ edge_index,
                            const int* __restrict__ edge_attr,
                            const float* __restrict__ bond_emb, // [L,3,16,D]
                            const float* __restrict__ h,
                            float* __restrict__ agg, int l) {
  int idx = blockIdx.x * blockDim.x + threadIdx.x;
  if (idx >= EE * (DD / 4)) return;
  int e = idx / (DD / 4), d4 = idx % (DD / 4);
  int src = edge_index[e];
  int dst = edge_index[EE + e];
  vf4 bv = {0.f, 0.f, 0.f, 0.f};
#pragma unroll
  for (int c = 0; c < 3; ++c) {
    int code = edge_attr[e * 3 + c];
    bv += *(const vf4*)(bond_emb + ((size_t)((l * 3 + c) * 16 + code)) * DD + d4 * 4);
  }
  vf4 hv = *(const vf4*)(h + (size_t)src * DD + d4 * 4);
  float* ap = agg + (size_t)dst * DD + d4 * 4;
#pragma unroll
  for (int j = 0; j < 4; ++j) {
    float m = fmaxf(hv[j] + bv[j], 0.f);
    atomicAdd(ap + j, m);
  }
}

// ---- z = (1+eps)*h + agg -> padded bf16 [N, KP1] ---------------------------
__global__ void z_cvt_kernel(const float* __restrict__ h,
                             const float* __restrict__ agg,
                             const float* __restrict__ eps, int l,
                             unsigned short* __restrict__ zbf) {
  int idx = blockIdx.x * blockDim.x + threadIdx.x;
  if (idx >= NN * (KP1 / 4)) return;
  int row = idx / (KP1 / 4), c4 = idx % (KP1 / 4);
  vu4 o = {0, 0, 0, 0};
  if (c4 < DD / 4) {
    float ep = 1.f + eps[l];
    vf4 hv = *(const vf4*)(h + (size_t)row * DD + c4 * 4);
    vf4 av = *(const vf4*)(agg + (size_t)row * DD + c4 * 4);
#pragma unroll
    for (int j = 0; j < 4; ++j) o[j] = f2bf(ep * hv[j] + av[j]);
  }
  *(vu4*)(zbf + (size_t)row * KP1 + c4 * 4) = o;
}

// ---- Weight repack: W[K,Nc] f32 -> Wt[NPad, KPad] bf16 (transposed) --------
__global__ void pack_wt_kernel(const float* __restrict__ W, int K, int Nc,
                               unsigned short* __restrict__ Wt, int KPad, int NPad) {
  int idx = blockIdx.x * blockDim.x + threadIdx.x;
  if (idx >= NPad * KPad) return;
  int n = idx / KPad, k = idx % KPad;
  float v = (k < K && n < Nc) ? W[(size_t)k * Nc + n] : 0.f;
  Wt[idx] = f2bf(v);
}

// ---- WMMA GEMM strip: one wave computes a 16(M) x 64(N) strip --------------
// A layout used here (contiguous k, not the ISA table order), 16-bit A 16x32:
//   lane L: M=L&15, klo=16*(L>>4); VGPR0..3=K[klo..klo+7], VGPR4..7=K[klo+8..klo+15]
// B (pre-transposed, N-major): lane L: Ncol=L&15, K contiguous per half-wave.
template <bool OUT_BF16, int TN>
__global__ void wmma_gemm_kernel(const unsigned short* __restrict__ A,
                                 const unsigned short* __restrict__ Bt,
                                 const float* __restrict__ bias, int biasN,
                                 void* __restrict__ out, int ldo,
                                 int K, int stripsN) {
  int wid = (blockIdx.x * blockDim.x + threadIdx.x) >> 5;
  int lane = threadIdx.x & 31;
  int sN = wid % stripsN;
  int tM = wid / stripsN;
  if (tM >= (NN >> 4)) return;   // uniform per wave -> whole-wave exit, EXEC stays full

  int m  = lane & 15;
  int hv = lane >> 4;
  const unsigned short* arow = A  + (size_t)(tM * 16 + m) * K + hv * 16;
  const unsigned short* bbase = Bt + ((size_t)(sN * TN) * 16 + m) * K + hv * 16;

  v8f c[TN];
#pragma unroll
  for (int t = 0; t < TN; ++t) c[t] = (v8f){};

  for (int kb = 0; kb < K; kb += 32) {
    v8bf alo = *(const v8bf*)(arow + kb);
    v8bf ahi = *(const v8bf*)(arow + kb + 8);
    v16bf a = __builtin_shufflevector(alo, ahi,
                                      0, 1, 2, 3, 4, 5, 6, 7,
                                      8, 9, 10, 11, 12, 13, 14, 15);
#pragma unroll
    for (int t = 0; t < TN; ++t) {
      v16bf b = *(const v16bf*)(bbase + (size_t)t * 16 * K + kb);
      c[t] = __builtin_amdgcn_wmma_f32_16x16x32_bf16(false, a, false, b,
                                                     (short)0, c[t], false, false);
    }
  }

#pragma unroll
  for (int t = 0; t < TN; ++t) {
    int col = (sN * TN + t) * 16 + m;
    float bv = (col < biasN) ? bias[col] : 0.f;
#pragma unroll
    for (int r = 0; r < 8; ++r) {
      int row = tM * 16 + r + hv * 8;
      float v = c[t][r] + bv;
      if (OUT_BF16)
        ((unsigned short*)out)[(size_t)row * ldo + col] = f2bf(v);
      else
        ((float*)out)[(size_t)row * ldo + col] = v;
    }
  }
}

// ---- Column mean / inv-std over rows ---------------------------------------
__global__ void col_stats_bf16_kernel(const unsigned short* __restrict__ in, int ld,
                                      float* __restrict__ mean, float* __restrict__ inv) {
  __shared__ float s1[256], s2[256];
  int col = blockIdx.x;
  float a = 0.f, b = 0.f;
  for (int r = threadIdx.x; r < NN; r += blockDim.x) {
    float v = bf2f(in[(size_t)r * ld + col]);
    a += v; b += v * v;
  }
  s1[threadIdx.x] = a; s2[threadIdx.x] = b;
  __syncthreads();
  for (int s = 128; s > 0; s >>= 1) {
    if (threadIdx.x < s) { s1[threadIdx.x] += s1[threadIdx.x + s]; s2[threadIdx.x] += s2[threadIdx.x + s]; }
    __syncthreads();
  }
  if (threadIdx.x == 0) {
    float mu = s1[0] / (float)NN;
    float var = s2[0] / (float)NN - mu * mu;
    mean[col] = mu;
    inv[col] = rsqrtf(var + 1e-5f);
  }
}

__global__ void col_stats_f32_kernel(const float* __restrict__ in, int ld,
                                     float* __restrict__ mean, float* __restrict__ inv) {
  __shared__ float s1[256], s2[256];
  int col = blockIdx.x;
  float a = 0.f, b = 0.f;
  for (int r = threadIdx.x; r < NN; r += blockDim.x) {
    float v = in[(size_t)r * ld + col];
    a += v; b += v * v;
  }
  s1[threadIdx.x] = a; s2[threadIdx.x] = b;
  __syncthreads();
  for (int s = 128; s > 0; s >>= 1) {
    if (threadIdx.x < s) { s1[threadIdx.x] += s1[threadIdx.x + s]; s2[threadIdx.x] += s2[threadIdx.x + s]; }
    __syncthreads();
  }
  if (threadIdx.x == 0) {
    float mu = s1[0] / (float)NN;
    float var = s2[0] / (float)NN - mu * mu;
    mean[col] = mu;
    inv[col] = rsqrtf(var + 1e-5f);
  }
}

// ---- Inner BN + ReLU, in-place (ushort4) on bf16 hidden [N, NP1] -----------
__global__ void bn_relu_bf16_kernel(unsigned short* __restrict__ hbuf,
                                    const float* __restrict__ mean,
                                    const float* __restrict__ inv,
                                    const float* __restrict__ g,
                                    const float* __restrict__ be) {
  int idx = blockIdx.x * blockDim.x + threadIdx.x;
  if (idx >= NN * (NP1 / 4)) return;
  int c4 = idx % (NP1 / 4);
  vu4* p = (vu4*)hbuf + idx;   // contiguous [N, NP1] layout
  if (c4 >= D2 / 4) { *p = (vu4){0, 0, 0, 0}; return; }
  vu4 xin = *p;
  vu4 o;
#pragma unroll
  for (int j = 0; j < 4; ++j) {
    int col = c4 * 4 + j;
    float x = bf2f(xin[j]);
    float y = (x - mean[col]) * inv[col] * g[col] + be[col];
    o[j] = f2bf(fmaxf(y, 0.f));
  }
  *p = o;
}

// ---- Outer BN (+optional ReLU): z2[N,NP2] f32 -> h[N,D] f32 (float4) -------
__global__ void bn_out_kernel(const float* __restrict__ z2,
                              const float* __restrict__ mean,
                              const float* __restrict__ inv,
                              const float* __restrict__ gamma,
                              const float* __restrict__ beta,
                              float* __restrict__ h, int do_relu) {
  int idx = blockIdx.x * blockDim.x + threadIdx.x;
  if (idx >= NN * (DD / 4)) return;
  int row = idx / (DD / 4), c4 = idx % (DD / 4);
  vf4 x = *(const vf4*)(z2 + (size_t)row * NP2 + c4 * 4);
  vf4 y;
#pragma unroll
  for (int j = 0; j < 4; ++j) {
    int col = c4 * 4 + j;
    float v = (x[j] - mean[col]) * inv[col] * gamma[col] + beta[col];
    y[j] = do_relu ? fmaxf(v, 0.f) : v;
  }
  *(vf4*)(h + (size_t)row * DD + c4 * 4) = y;
}

// ---------------------------------------------------------------------------
extern "C" void kernel_launch(void* const* d_in, const int* in_sizes, int n_in,
                              void* d_out, int out_size, void* d_ws, size_t ws_size,
                              hipStream_t stream) {
  const int*   x          = (const int*)d_in[0];
  const int*   edge_index = (const int*)d_in[1];
  const int*   edge_attr  = (const int*)d_in[2];
  const float* atom_emb   = (const float*)d_in[3];
  const float* bond_emb   = (const float*)d_in[4];
  const float* eps        = (const float*)d_in[5];
  const float* W1         = (const float*)d_in[6];
  const float* b1         = (const float*)d_in[7];
  const float* g1         = (const float*)d_in[8];
  const float* be1        = (const float*)d_in[9];
  const float* W2         = (const float*)d_in[10];
  const float* b2         = (const float*)d_in[11];
  const float* gamma      = (const float*)d_in[12];
  const float* beta       = (const float*)d_in[13];

  float* h = (float*)d_out;   // persistent node features live in d_out

  size_t off = 0;
  auto carve = [&](size_t bytes) -> void* {
    void* p = (char*)d_ws + off;
    off += (bytes + 255) & ~(size_t)255;
    return p;
  };
  float*          agg   = (float*)carve((size_t)NN * DD * 4);
  unsigned short* zbf   = (unsigned short*)carve((size_t)NN * KP1 * 2);
  unsigned short* h1bf  = (unsigned short*)carve((size_t)NN * NP1 * 2);
  float*          z2    = (float*)carve((size_t)NN * NP2 * 4);
  unsigned short* W1t   = (unsigned short*)carve((size_t)NP1 * KP1 * 2);
  unsigned short* W2t   = (unsigned short*)carve((size_t)NP2 * KP2 * 2);
  float*          mean1 = (float*)carve(NP1 * 4);
  float*          inv1  = (float*)carve(NP1 * 4);
  float*          mean2 = (float*)carve(NP2 * 4);
  float*          inv2  = (float*)carve(NP2 * 4);

  const int TB = 256;
  auto blocks = [](long long tot, int tb) { return (unsigned)((tot + tb - 1) / tb); };

  atom_encoder_kernel<<<blocks((long long)NN * (DD / 4), TB), TB, 0, stream>>>(x, atom_emb, h);

  const int tilesM   = NN / 16;           // 3125
  const int stripsN1 = NP1 / 64;          // 10
  const int stripsN2 = NP2 / 64;          // 5

  for (int l = 0; l < LL; ++l) {
    // GIN aggregate
    hipMemsetAsync(agg, 0, (size_t)NN * DD * 4, stream);
    edge_kernel<<<blocks((long long)EE * (DD / 4), TB), TB, 0, stream>>>(
        edge_index, edge_attr, bond_emb, h, agg, l);

    // z = (1+eps)h + agg, bf16-packed for GEMM1
    z_cvt_kernel<<<blocks((long long)NN * (KP1 / 4), TB), TB, 0, stream>>>(h, agg, eps, l, zbf);

    // GEMM1: [N,320] x [320,640] -> h1bf (bf16, bias fused)
    pack_wt_kernel<<<blocks((long long)NP1 * KP1, TB), TB, 0, stream>>>(
        W1 + (size_t)l * DD * D2, DD, D2, W1t, KP1, NP1);
    {
      long long waves = (long long)tilesM * stripsN1;
      wmma_gemm_kernel<true, 4><<<blocks(waves * 32, TB), TB, 0, stream>>>(
          zbf, W1t, b1 + (size_t)l * D2, D2, (void*)h1bf, NP1, KP1, stripsN1);
    }

    // inner BN + ReLU (in place on bf16)
    col_stats_bf16_kernel<<<D2, 256, 0, stream>>>(h1bf, NP1, mean1, inv1);
    bn_relu_bf16_kernel<<<blocks((long long)NN * (NP1 / 4), TB), TB, 0, stream>>>(
        h1bf, mean1, inv1, g1 + (size_t)l * D2, be1 + (size_t)l * D2);

    // GEMM2: [N,640] x [640,320] -> z2 (f32, bias fused)
    pack_wt_kernel<<<blocks((long long)NP2 * KP2, TB), TB, 0, stream>>>(
        W2 + (size_t)l * D2 * DD, D2, DD, W2t, KP2, NP2);
    {
      long long waves = (long long)tilesM * stripsN2;
      wmma_gemm_kernel<false, 4><<<blocks(waves * 32, TB), TB, 0, stream>>>(
          h1bf, W2t, b2 + (size_t)l * DD, DD, (void*)z2, NP2, KP2, stripsN2);
    }

    // outer BN (+ReLU except last layer) -> h
    col_stats_f32_kernel<<<DD, 256, 0, stream>>>(z2, NP2, mean2, inv2);
    bn_out_kernel<<<blocks((long long)NN * (DD / 4), TB), TB, 0, stream>>>(
        z2, mean2, inv2, gamma + (size_t)l * DD, beta + (size_t)l * DD, h,
        (l < LL - 1) ? 1 : 0);
  }
}
